// RTFMVDRBeamformer_55748675502291
// MI455X (gfx1250) — matrix operand order repaired and checked by reading, not compiled, not run on hardware
//
#include <hip/hip_runtime.h>

#define Bc 16
#define Mc 8
#define Fc 257
#define Tc 1000

typedef __attribute__((ext_vector_type(16))) _Float16 v16h;
typedef __attribute__((ext_vector_type(8)))  float    v8f;

// ---------------------------------------------------------------------------
// Kernel 1: per-(b,f) MVDR weight solve.
//   - power iteration on target SCM -> dominant eigenvector a (rtf)
//   - complex Cholesky of noise SCM, solve N x = a
//   - w = x / (a^H x)   (denominator is real for Hermitian PSD N)
// One thread per (b,f); loads coalesce across the wave since f is the
// contiguous thread dimension and the SCM layout is (B,M,M,F,2).
// ---------------------------------------------------------------------------
__global__ __launch_bounds__(128) void mvdr_weights_kernel(
    const float* __restrict__ tscm, const float* __restrict__ nscm,
    float* __restrict__ wbuf) {
  int g = blockIdx.x * blockDim.x + threadIdx.x;
  if (g >= Bc * Fc) return;
  int b = g / Fc, f = g % Fc;

  float ar[Mc][Mc], ai[Mc][Mc];
#pragma unroll
  for (int i = 0; i < Mc; ++i)
#pragma unroll
    for (int j = 0; j < Mc; ++j) {
      size_t off = ((((size_t)b * Mc + i) * Mc + j) * Fc + (size_t)f) * 2;
      ar[i][j] = tscm[off];
      ai[i][j] = tscm[off + 1];
    }

  // Power iteration for dominant eigenvector (PSD -> converges to eigh[...,-1])
  float vr[Mc], vi[Mc];
#pragma unroll
  for (int m = 0; m < Mc; ++m) { vr[m] = 1.0f; vi[m] = 0.0f; }
  for (int it = 0; it < 24; ++it) {
    float ur[Mc], ui[Mc];
#pragma unroll
    for (int i = 0; i < Mc; ++i) {
      float sr = 0.0f, si = 0.0f;
#pragma unroll
      for (int j = 0; j < Mc; ++j) {
        sr += ar[i][j] * vr[j] - ai[i][j] * vi[j];
        si += ar[i][j] * vi[j] + ai[i][j] * vr[j];
      }
      ur[i] = sr; ui[i] = si;
    }
    float nn = 0.0f;
#pragma unroll
    for (int i = 0; i < Mc; ++i) nn += ur[i] * ur[i] + ui[i] * ui[i];
    float inv = rsqrtf(fmaxf(nn, 1e-30f));
#pragma unroll
    for (int i = 0; i < Mc; ++i) { vr[i] = ur[i] * inv; vi[i] = ui[i] * inv; }
  }

  // Reload registers with noise SCM
#pragma unroll
  for (int i = 0; i < Mc; ++i)
#pragma unroll
    for (int j = 0; j < Mc; ++j) {
      size_t off = ((((size_t)b * Mc + i) * Mc + j) * Fc + (size_t)f) * 2;
      ar[i][j] = nscm[off];
      ai[i][j] = nscm[off + 1];
    }

  // In-place complex Cholesky (lower): N = L L^H
#pragma unroll
  for (int j = 0; j < Mc; ++j) {
    float d = ar[j][j];
#pragma unroll
    for (int k = 0; k < j; ++k) d -= ar[j][k] * ar[j][k] + ai[j][k] * ai[j][k];
    d = fmaxf(d, 1e-12f);
    float s = sqrtf(d);
    float invs = 1.0f / s;
    ar[j][j] = s; ai[j][j] = 0.0f;
#pragma unroll
    for (int i = j + 1; i < Mc; ++i) {
      float cr = ar[i][j], ci = ai[i][j];
#pragma unroll
      for (int k = 0; k < j; ++k) {
        cr -= ar[i][k] * ar[j][k] + ai[i][k] * ai[j][k];
        ci -= ai[i][k] * ar[j][k] - ar[i][k] * ai[j][k];
      }
      ar[i][j] = cr * invs;
      ai[i][j] = ci * invs;
    }
  }

  // Forward solve L y = a
  float yr[Mc], yi[Mc];
#pragma unroll
  for (int i = 0; i < Mc; ++i) {
    float sr = vr[i], si = vi[i];
#pragma unroll
    for (int k = 0; k < i; ++k) {
      sr -= ar[i][k] * yr[k] - ai[i][k] * yi[k];
      si -= ar[i][k] * yi[k] + ai[i][k] * yr[k];
    }
    float inv = 1.0f / ar[i][i];
    yr[i] = sr * inv; yi[i] = si * inv;
  }
  // Back solve L^H x = y : uses conj(L[k][i])
  float xr[Mc], xi[Mc];
#pragma unroll
  for (int i = Mc - 1; i >= 0; --i) {
    float sr = yr[i], si = yi[i];
#pragma unroll
    for (int k = i + 1; k < Mc; ++k) {
      sr -= ar[k][i] * xr[k] + ai[k][i] * xi[k];
      si -= ar[k][i] * xi[k] - ai[k][i] * xr[k];
    }
    float inv = 1.0f / ar[i][i];
    xr[i] = sr * inv; xi[i] = si * inv;
  }

  // denom = a^H N^{-1} a  (real, positive)
  float den = 0.0f;
#pragma unroll
  for (int m = 0; m < Mc; ++m) den += vr[m] * xr[m] + vi[m] * xi[m];
  float invd = 1.0f / den;

  float* wp = wbuf + (size_t)g * 16;
#pragma unroll
  for (int m = 0; m < Mc; ++m) {
    wp[m]     = xr[m] * invd;  // w_r
    wp[8 + m] = xi[m] * invd;  // w_i
  }
}

// ---------------------------------------------------------------------------
// Kernel 2: apply beamformer via WMMA.
//   out_re(t) = sum_m w_r x_r + w_i x_i ; out_im(t) = sum_m w_r x_i - w_i x_r
// Real GEMM D(2x16) = A(2x16, K padded to 32) * X(16x16t) per 16-wide t tile,
// using v_wmma_f32_16x16x32_f16 with hi/lo f16 split (3 chained WMMAs) for
// near-fp32 accuracy. One wave per (b,f,T-half). Memory-bound: ~300 MB total.
// ---------------------------------------------------------------------------
__global__ __launch_bounds__(256) void mvdr_apply_kernel(
    const float* __restrict__ mix, const float* __restrict__ wbuf,
    float* __restrict__ out) {
  const int SEG = 2;  // waves per (b,f), each covers half of T
  int gwave = (int)((blockIdx.x * blockDim.x + threadIdx.x) >> 5);
  int lane = threadIdx.x & 31;
  int bf = gwave / SEG;
  int seg = gwave % SEG;
  if (bf >= Bc * Fc) return;
  int b = bf / Fc, f = bf % Fc;

  const float* wp = wbuf + (size_t)bf * 16;

  // A fragments (this code's K order, same as B): lane<16 holds row M=lane, K=0..15 in
  // halves[0..15]; lanes>=16 hold K=16..31 = 0 pad.
  // Row0 = [w_r | w_i], Row1 = [-w_i | w_r]; all other rows zero.
  v16h a_hi = {};
  v16h a_lo = {};
  int r = lane & 15;
  bool upper = lane >= 16;
  if (r < 2 && !upper) {
#pragma unroll
    for (int m = 0; m < 8; ++m) {
      float wr = wp[m], wi = wp[8 + m];
      float av = (r == 0) ? wr : -wi;
      float bv = (r == 0) ? wi : wr;
      _Float16 h = (_Float16)av;
      _Float16 h2 = (_Float16)bv;
      a_hi[m] = h;
      a_lo[m] = (_Float16)(av - (float)h);
      a_hi[8 + m] = h2;
      a_lo[8 + m] = (_Float16)(bv - (float)h2);
    }
  }

  const size_t mstride = (size_t)Fc * Tc * 2;  // stride between mics (floats)
  const float* mixbf = mix + ((size_t)b * Mc * Fc + (size_t)f) * (size_t)Tc * 2;

  int t_begin = seg * 512;
  int t_end = t_begin + 512;
  if (t_end > Tc) t_end = Tc;

  for (int t0 = t_begin; t0 < t_end; t0 += 16) {
    // B fragment (32x16 f16): lane<16 = column t0+lane, halves[0..15] =
    // K=0..15 = [x_r(m=0..7), x_i(m=0..7)]; lanes 16..31 = K=16..31 = 0 pad.
    v16h xh = {};
    v16h xl = {};
    int t = t0 + r;
    bool active = (!upper) && (t < Tc);
    if (active) {
#pragma unroll
      for (int m = 0; m < 8; ++m) {
        const float* p = mixbf + (size_t)m * mstride + (size_t)t * 2;
        float xre = p[0];
        float xim = p[1];
        _Float16 hre = (_Float16)xre;
        _Float16 him = (_Float16)xim;
        xh[m]     = hre;
        xh[8 + m] = him;
        xl[m]     = (_Float16)(xre - (float)hre);
        xl[8 + m] = (_Float16)(xim - (float)him);
      }
    }

    // Stream prefetch ~4 tiles (64 t) ahead: lanes fan out over mics/offsets.
    {
      int tp = t0 + 64;
      if (tp < t_end) {
        const float* pp = mixbf + (size_t)(lane & 7) * mstride +
                          (size_t)(tp + ((lane >> 3) << 2)) * 2;
        __builtin_prefetch(pp, 0, 1);
      }
    }

    // D = A_lo*X_hi + A_hi*X_lo + A_hi*X_hi  (f32 accumulate)
    v8f acc = {};
    acc = __builtin_amdgcn_wmma_f32_16x16x32_f16(false, a_lo, false, xh,
                                                 (short)0, acc, false, false);
    acc = __builtin_amdgcn_wmma_f32_16x16x32_f16(false, a_hi, false, xl,
                                                 (short)0, acc, false, false);
    acc = __builtin_amdgcn_wmma_f32_16x16x32_f16(false, a_hi, false, xh,
                                                 (short)0, acc, false, false);

    // D layout: lane n (n<16): acc[0] = D[M=0][n] = re(t0+n), acc[1] = im.
    if (active) {
      float2 o;
      o.x = acc[0];
      o.y = acc[1];
      *(float2*)(out + ((size_t)bf * Tc + (size_t)t) * 2) = o;
    }
  }
}

// ---------------------------------------------------------------------------
extern "C" void kernel_launch(void* const* d_in, const int* in_sizes, int n_in,
                              void* d_out, int out_size, void* d_ws,
                              size_t ws_size, hipStream_t stream) {
  const float* mix  = (const float*)d_in[0];  // (B,M,F,T,2) f32
  const float* tscm = (const float*)d_in[1];  // (B,M,M,F,2) f32
  const float* nscm = (const float*)d_in[2];  // (B,M,M,F,2) f32
  float* out  = (float*)d_out;                // (B,F,T) complex64 interleaved
  float* wbuf = (float*)d_ws;                 // 4112 * 16 floats = 263 KB

  const int BF = Bc * Fc;  // 4112
  mvdr_weights_kernel<<<(BF + 127) / 128, 128, 0, stream>>>(tscm, nscm, wbuf);

  // 2 waves per (b,f) -> 8224 waves; 8 waves (256 thr) per block -> 1028 blocks
  const int waves = BF * 2;
  mvdr_apply_kernel<<<waves / 8, 256, 0, stream>>>(mix, wbuf, out);
}
